// Seq2Seq_53171695124990
// MI455X (gfx1250) — matrix operand order repaired and checked by reading, not compiled, not run on hardware
//
#include <hip/hip_runtime.h>
#include <hip/hip_bf16.h>
#include <stdint.h>

// Sizes from the reference
#define Vv   50000
#define Dd   300
#define Dp   320      // D padded to multiple of 32 for bf16 WMMA K
#define Hh   512
#define Bb   32
#define LCc  400
#define LQm1 29
#define NEGV (-1e30f)

typedef unsigned short u16;
typedef __attribute__((ext_vector_type(16))) __bf16 v16bf;
typedef __attribute__((ext_vector_type(8)))  float  v8f;

__device__ __forceinline__ int imin(int a, int b) { return a < b ? a : b; }

__device__ __forceinline__ u16 f2bf(float f) {
  union { float f; uint32_t u; } x; x.f = f;
  uint32_t r = (x.u + 0x7FFFu + ((x.u >> 16) & 1u)) >> 16;  // RNE
  return (u16)r;
}
__device__ __forceinline__ float bf2f(u16 b) {
  union { uint32_t u; float f; } x; x.u = ((uint32_t)b) << 16;
  return x.f;
}
__device__ __forceinline__ float sigm(float x) { return 1.f / (1.f + __expf(-x)); }

union BFrag { v16bf v; u16 u[16]; uint4 q[2]; };

__device__ __forceinline__ v8f wmma_bf16(v16bf a, v16bf b, v8f c) {
  // D = A(16x32 bf16) * B(32x16 bf16) + C(16x16 f32)
  return __builtin_amdgcn_wmma_f32_16x16x32_bf16(false, a, false, b, (short)0, c, false, false);
}

// A fragment (16x32 bf16) from row-major bf16 [*, lda]; rows m0..m0+15, k = kk..kk+31.
// ISA layout: lanes 0-15 hold row m, K 0..7 & 16..23; lanes 16-31 hold row m, K 8..15 & 24..31.
__device__ __forceinline__ v16bf load_a_bf16(const u16* A, int lda, int m0, int kk, int lane) {
  BFrag f;
  const u16* p = A + (long)(m0 + (lane & 15)) * lda + kk + ((lane >> 4) << 3);
  f.q[0] = *(const uint4*)(p);
  f.q[1] = *(const uint4*)(p + 16);
  return f.v;
}

// B fragment (32x16 bf16) from row-major bf16 W [N, ldk]; B(k,n) = W[n,k].
// ISA layout: lanes 0-15 -> col n, K 0..7 & 16..23; lanes 16-31 -> col n, K 8..15 & 24..31.
__device__ __forceinline__ v16bf load_b_bf16(const u16* W, int ldk, int n0, int kk, int lane) {
  BFrag f;
  const u16* p = W + (long)(n0 + (lane & 15)) * ldk + kk + ((lane >> 4) << 3);
  f.q[0] = *(const uint4*)(p);
  f.q[1] = *(const uint4*)(p + 16);
  return f.v;
}

// A fragment gathered on the fly from an fp32 row, zero-padded past klim.
__device__ __forceinline__ v16bf load_a_f32row(const float* row, int kk, int klim, int lane) {
  BFrag f;
  int sel = (lane >> 4) << 3;
#pragma unroll
  for (int i = 0; i < 8; ++i) {
    int k0 = kk + sel + i;
    int k1 = kk + 16 + sel + i;
    f.u[i]     = (k0 < klim) ? f2bf(row[k0]) : (u16)0;
    f.u[8 + i] = (k1 < klim) ? f2bf(row[k1]) : (u16)0;
  }
  return f.v;
}

// A fragment from concatenated [h ; x] (K = 1024 = Hh + Hh); kk is compile-time after unroll.
__device__ __forceinline__ v16bf load_a_hx(const u16* h, const u16* x, int ldx,
                                           int m0, int kk, int lane) {
  return (kk < Hh) ? load_a_bf16(h, Hh, m0, kk, lane)
                   : load_a_bf16(x, ldx, m0, kk - Hh, lane);
}

// ---------------- prep kernels (run once per call) ----------------

// Concatenate [Whh ; Wih] -> Wcat [2048, 1024] bf16, gbias = bih + bhh.
__global__ __launch_bounds__(256) void k_prep_w(const float* __restrict__ Whh,
                                                const float* __restrict__ Wih,
                                                const float* __restrict__ bih,
                                                const float* __restrict__ bhh,
                                                u16* __restrict__ Wcat,
                                                float* __restrict__ gbias) {
  int n = blockIdx.x;  // 0..2047
  for (int k = threadIdx.x; k < 2 * Hh; k += 256)
    Wcat[(long)n * (2 * Hh) + k] =
        f2bf(k < Hh ? Whh[(long)n * Hh + k] : Wih[(long)n * Hh + (k - Hh)]);
  if (threadIdx.x == 0) gbias[n] = bih[n] + bhh[n];
}

// emb_W [512,300] fp32 -> [512,320] bf16 zero-padded.
__global__ __launch_bounds__(256) void k_prep_embw(const float* __restrict__ embW,
                                                   u16* __restrict__ Wb) {
  int h = blockIdx.x;  // 0..511
  for (int k = threadIdx.x; k < Dp; k += 256)
    Wb[(long)h * Dp + k] = (k < Dd) ? f2bf(embW[(long)h * Dd + k]) : (u16)0;
}

// fp32 -> bf16, 8 elements/thread (float4 x2 in, uint4 out); n must be multiple of 8.
__global__ __launch_bounds__(256) void k_cvt_bf16x8(const float* __restrict__ src,
                                                    u16* __restrict__ dst, long n8) {
  long i = (long)blockIdx.x * 256 + threadIdx.x;
  if (i < n8) {
    const float4* s = (const float4*)src;
    float4 x0 = s[2 * i], x1 = s[2 * i + 1];
    union { uint4 q; u16 u[8]; } o;
    o.u[0] = f2bf(x0.x); o.u[1] = f2bf(x0.y); o.u[2] = f2bf(x0.z); o.u[3] = f2bf(x0.w);
    o.u[4] = f2bf(x1.x); o.u[5] = f2bf(x1.y); o.u[6] = f2bf(x1.z); o.u[7] = f2bf(x1.w);
    ((uint4*)dst)[i] = o.q;
  }
}

// c_len, query tokens/mask, zero LSTM state.
__global__ __launch_bounds__(256) void k_meta(const int* __restrict__ cw,
                                              const int* __restrict__ qw,
                                              int* __restrict__ c_len,
                                              int* __restrict__ qtok,
                                              int* __restrict__ qmask,
                                              u16* __restrict__ h_a, u16* __restrict__ h_b,
                                              float* __restrict__ c_st) {
  for (int b = threadIdx.x; b < Bb; b += 256) {
    int cnt = 0;
    for (int t = 0; t < LCc; ++t) cnt += (cw[b * LCc + t] != 0);
    c_len[b] = cnt;
  }
  for (int i = threadIdx.x; i < Bb * LQm1; i += 256) {
    int b = i / LQm1, t = i % LQm1;
    int tk = qw[b * (LQm1 + 1) + t];  // qw_idxs[:, :-1]
    qtok[i] = tk;
    qmask[i] = (tk != 0);
  }
  for (int i = threadIdx.x; i < Bb * Hh; i += 256) {
    h_a[i] = 0; h_b[i] = 0; c_st[i] = 0.f;
  }
}

// ---------------- embedding GEMM: out[m, 0:512] = wv[tok[m]] @ embW^T ----------------
// Block = one 16-row m-tile, 8 waves x 4 n-tiles = 512 cols.
__global__ __launch_bounds__(256) void k_embed(const int* __restrict__ tok,
                                               const float* __restrict__ wv,
                                               const u16* __restrict__ Wb,  // [512,320] bf16
                                               u16* __restrict__ out) {    // [M,512] bf16
  int w = threadIdx.x >> 5, lane = threadIdx.x & 31;
  int m0 = blockIdx.x * 16;
  const float* row = wv + (long)tok[m0 + (lane & 15)] * Dd;
  v8f acc[4];
#pragma unroll
  for (int nt = 0; nt < 4; ++nt)
#pragma unroll
    for (int r = 0; r < 8; ++r) acc[nt][r] = 0.f;

#pragma unroll
  for (int kk = 0; kk < Dp; kk += 32) {
    v16bf a = load_a_f32row(row, kk, Dd, lane);
#pragma unroll
    for (int nt = 0; nt < 4; ++nt) {
      v16bf b = load_b_bf16(Wb, Dp, w * 64 + nt * 16, kk, lane);
      acc[nt] = wmma_bf16(a, b, acc[nt]);
    }
  }
  int hi = lane >> 4, col = lane & 15;
#pragma unroll
  for (int nt = 0; nt < 4; ++nt)
#pragma unroll
    for (int r = 0; r < 8; ++r)
      out[(long)(m0 + hi * 8 + r) * Hh + w * 64 + nt * 16 + col] = f2bf(acc[nt][r]);
}

// ---------------- one LSTM step ----------------
// gates[32,2048] = gbias + [h ; x_t] @ Wcat^T   (Wcat = [Whh;Wih], K=1024, bf16 WMMA)
// 8 blocks x 64 cols; block: 8 waves -> (gate g, m-tile) x 4 n-tiles; update via LDS.
// Fully unrolled, software-pipelined k-loop (loads of chunk ch+1 overlap WMMA of ch).
__global__ __launch_bounds__(256) void k_lstm_step(const u16* __restrict__ h_in,
                                                   u16* __restrict__ h_out,
                                                   float* __restrict__ c_state,
                                                   const u16* __restrict__ xbase, int ldx,
                                                   const u16* __restrict__ Wcat,
                                                   const float* __restrict__ gbias,
                                                   const int* __restrict__ c_len, int t,
                                                   u16* __restrict__ dec_out) {
  __shared__ float gbuf[4][Bb][64 + 1];
  int w = threadIdx.x >> 5, lane = threadIdx.x & 31;
  int g = w >> 1, mt = w & 1, m0 = mt * 16;
  int j0 = blockIdx.x * 64;
  int nbase = g * Hh + j0;

  v8f acc[4];
#pragma unroll
  for (int nt = 0; nt < 4; ++nt) {
    float bv = gbias[nbase + nt * 16 + (lane & 15)];
#pragma unroll
    for (int r = 0; r < 8; ++r) acc[nt][r] = bv;
  }

  constexpr int NCH = (2 * Hh) / 32;  // 32 chunks
  v16bf a_cur = load_a_hx(h_in, xbase, ldx, m0, 0, lane);
  v16bf b_cur[4];
#pragma unroll
  for (int nt = 0; nt < 4; ++nt) b_cur[nt] = load_b_bf16(Wcat, 2 * Hh, nbase + nt * 16, 0, lane);

#pragma unroll
  for (int ch = 0; ch < NCH - 1; ++ch) {
    int kn = (ch + 1) * 32;
    v16bf a_nxt = load_a_hx(h_in, xbase, ldx, m0, kn, lane);
    v16bf b_nxt[4];
#pragma unroll
    for (int nt = 0; nt < 4; ++nt)
      b_nxt[nt] = load_b_bf16(Wcat, 2 * Hh, nbase + nt * 16, kn, lane);
#pragma unroll
    for (int nt = 0; nt < 4; ++nt) acc[nt] = wmma_bf16(a_cur, b_cur[nt], acc[nt]);
    a_cur = a_nxt;
#pragma unroll
    for (int nt = 0; nt < 4; ++nt) b_cur[nt] = b_nxt[nt];
  }
#pragma unroll
  for (int nt = 0; nt < 4; ++nt) acc[nt] = wmma_bf16(a_cur, b_cur[nt], acc[nt]);

  int hi = lane >> 4, col = lane & 15;
#pragma unroll
  for (int nt = 0; nt < 4; ++nt)
#pragma unroll
    for (int r = 0; r < 8; ++r)
      gbuf[g][m0 + hi * 8 + r][nt * 16 + col] = acc[nt][r];
  __syncthreads();

  for (int e = threadIdx.x; e < Bb * 64; e += 256) {
    int b = e >> 6, j = e & 63, jj = j0 + j;
    float ig = sigm(gbuf[0][b][j]);
    float fg = sigm(gbuf[1][b][j]);
    float gg = tanhf(gbuf[2][b][j]);
    float og = sigm(gbuf[3][b][j]);
    float c_old = c_state[b * Hh + jj];
    float nc = fg * c_old + ig * gg;
    float nh = og * tanhf(nc);
    if (c_len && !(t < c_len[b])) {  // freeze state past true length
      nc = c_old;
      nh = bf2f(h_in[b * Hh + jj]);
    }
    c_state[b * Hh + jj] = nc;
    h_out[b * Hh + jj] = f2bf(nh);
    if (dec_out) dec_out[(long)(b * LQm1 + t) * Hh + jj] = f2bf(nh);
  }
}

// ---------------- projection: logits[m, v] (+bias, mask) ----------------
// Wave = 4 m-tiles x 2 n-tiles (B fragments reused across 4 WMMAs).
// Block = 8 waves -> 16 n-tiles x 64 rows. grid = (196 n-blocks, 15 m-blocks):
// proj_W is read 15x (vs 58x with one m-tile/block) -> ~0.77 GB L2 traffic.
// Tile indices clamped (not branched) so EXEC stays all-ones for WMMA.
__global__ __launch_bounds__(256) void k_proj(const u16* __restrict__ A,   // [928,512] bf16
                                              const u16* __restrict__ Wp,  // [50000,512] bf16
                                              const float* __restrict__ bias,
                                              const int* __restrict__ qmask,
                                              float* __restrict__ out) {
  const int NT = Vv / 16;           // 3125 n-tiles
  const int MT = (Bb * LQm1) / 16;  // 58 m-tiles
  int w = threadIdx.x >> 5, lane = threadIdx.x & 31;
  int col = lane & 15, hi = lane >> 4;
  int ntb = blockIdx.x * 16 + w * 2;
  int mtb = blockIdx.y * 4;
  int nt0 = imin(ntb, NT - 1), nt1 = imin(ntb + 1, NT - 1);
  int mtc[4];
#pragma unroll
  for (int i = 0; i < 4; ++i) mtc[i] = imin(mtb + i, MT - 1);

  v8f acc[4][2];
  float b0 = bias[nt0 * 16 + col], b1 = bias[nt1 * 16 + col];
#pragma unroll
  for (int i = 0; i < 4; ++i)
#pragma unroll
    for (int r = 0; r < 8; ++r) { acc[i][0][r] = b0; acc[i][1][r] = b1; }

  constexpr int NCH = Hh / 32;  // 16 chunks
  v16bf aC[4], bC[2];
#pragma unroll
  for (int i = 0; i < 4; ++i) aC[i] = load_a_bf16(A, Hh, mtc[i] * 16, 0, lane);
  bC[0] = load_b_bf16(Wp, Hh, nt0 * 16, 0, lane);
  bC[1] = load_b_bf16(Wp, Hh, nt1 * 16, 0, lane);

#pragma unroll
  for (int ch = 0; ch < NCH - 1; ++ch) {
    int kn = (ch + 1) * 32;
    if (kn + 64 < Hh) {  // stream proj_W ahead through the cache hierarchy
      __builtin_prefetch((const void*)(Wp + (long)(nt0 * 16 + col) * Hh + kn + 64), 0, 0);
      __builtin_prefetch((const void*)(Wp + (long)(nt1 * 16 + col) * Hh + kn + 64), 0, 0);
    }
    v16bf bN[2];
    bN[0] = load_b_bf16(Wp, Hh, nt0 * 16, kn, lane);
    bN[1] = load_b_bf16(Wp, Hh, nt1 * 16, kn, lane);
    v16bf aN[4];
#pragma unroll
    for (int i = 0; i < 4; ++i) aN[i] = load_a_bf16(A, Hh, mtc[i] * 16, kn, lane);
#pragma unroll
    for (int i = 0; i < 4; ++i) {
      acc[i][0] = wmma_bf16(aC[i], bC[0], acc[i][0]);
      acc[i][1] = wmma_bf16(aC[i], bC[1], acc[i][1]);
    }
    bC[0] = bN[0]; bC[1] = bN[1];
#pragma unroll
    for (int i = 0; i < 4; ++i) aC[i] = aN[i];
  }
#pragma unroll
  for (int i = 0; i < 4; ++i) {
    acc[i][0] = wmma_bf16(aC[i], bC[0], acc[i][0]);
    acc[i][1] = wmma_bf16(aC[i], bC[1], acc[i][1]);
  }

#pragma unroll
  for (int i = 0; i < 4; ++i) {
    if (mtb + i >= MT) continue;  // tail m-tiles were redundant
#pragma unroll
    for (int r = 0; r < 8; ++r) {
      int m = mtc[i] * 16 + hi * 8 + r;
      bool keep = qmask[m] != 0;
      if (ntb < NT)     out[(long)m * Vv + nt0 * 16 + col] = keep ? acc[i][0][r] : NEGV;
      if (ntb + 1 < NT) out[(long)m * Vv + nt1 * 16 + col] = keep ? acc[i][1][r] : NEGV;
    }
  }
}

// ---------------- logsumexp per row (online max/sum, float4 reads) ----------------
__global__ __launch_bounds__(256) void k_lse(const float* __restrict__ out,
                                             float* __restrict__ logZ) {
  int row = blockIdx.x;
  const float4* p4 = (const float4*)(out + (long)row * Vv);
  float m = -INFINITY, s = 0.f;
  for (int v = threadIdx.x; v < Vv / 4; v += 256) {
    float4 x4 = p4[v];
    float xs[4] = {x4.x, x4.y, x4.z, x4.w};
#pragma unroll
    for (int i = 0; i < 4; ++i) {
      float x = xs[i];
      if (x > m) { s = s * __expf(m - x) + 1.f; m = x; }
      else       { s += __expf(x - m); }
    }
  }
  __shared__ float sm[256], ss[256];
  sm[threadIdx.x] = m; ss[threadIdx.x] = s;
  for (int off = 128; off > 0; off >>= 1) {
    __syncthreads();
    if (threadIdx.x < off) {
      float m1 = sm[threadIdx.x], s1 = ss[threadIdx.x];
      float m2 = sm[threadIdx.x + off], s2 = ss[threadIdx.x + off];
      float M = fmaxf(m1, m2);
      sm[threadIdx.x] = M;
      ss[threadIdx.x] = s1 * __expf(m1 - M) + s2 * __expf(m2 - M);
    }
  }
  __syncthreads();
  if (threadIdx.x == 0) logZ[row] = sm[0] + __logf(ss[0]);
}

__global__ __launch_bounds__(256) void k_norm(float* __restrict__ out,
                                              const float* __restrict__ logZ) {
  int row = blockIdx.y;
  int i = blockIdx.x * 256 + threadIdx.x;
  if (i < Vv / 4) {
    float z = logZ[row];
    float4* p = (float4*)(out + (long)row * Vv);
    float4 x = p[i];
    x.x -= z; x.y -= z; x.z -= z; x.w -= z;
    p[i] = x;
  }
}

// ---------------- launcher ----------------
extern "C" void kernel_launch(void* const* d_in, const int* in_sizes, int n_in,
                              void* d_out, int out_size, void* d_ws, size_t ws_size,
                              hipStream_t stream) {
  const int*   cw   = (const int*)d_in[0];
  const int*   qw   = (const int*)d_in[1];
  const float* wv   = (const float*)d_in[2];
  const float* embW = (const float*)d_in[3];
  const float* eWih = (const float*)d_in[4];
  const float* eWhh = (const float*)d_in[5];
  const float* ebih = (const float*)d_in[6];
  const float* ebhh = (const float*)d_in[7];
  const float* dWih = (const float*)d_in[8];
  const float* dWhh = (const float*)d_in[9];
  const float* dbih = (const float*)d_in[10];
  const float* dbhh = (const float*)d_in[11];
  const float* pW   = (const float*)d_in[12];
  const float* pb   = (const float*)d_in[13];
  float* out = (float*)d_out;

  char* wp = (char*)d_ws;
  auto take = [&](size_t bytes) -> void* {
    void* r = (void*)wp; wp += (bytes + 255) & ~(size_t)255; return r;
  };
  u16*   Wcat_e = (u16*)take((size_t)4 * Hh * 2 * Hh * 2);
  u16*   Wcat_d = (u16*)take((size_t)4 * Hh * 2 * Hh * 2);
  float* gb_e   = (float*)take(4 * Hh * 4);
  float* gb_d   = (float*)take(4 * Hh * 4);
  u16*   embWb  = (u16*)take((size_t)Hh * Dp * 2);
  u16*   projWb = (u16*)take((size_t)Vv * Hh * 2);
  u16*   c_emb  = (u16*)take((size_t)Bb * LCc * Hh * 2);
  u16*   q_emb  = (u16*)take((size_t)Bb * LQm1 * Hh * 2);
  u16*   h_a    = (u16*)take((size_t)Bb * Hh * 2);
  u16*   h_b    = (u16*)take((size_t)Bb * Hh * 2);
  float* c_st   = (float*)take((size_t)Bb * Hh * 4);
  u16*   dec_o  = (u16*)take((size_t)Bb * LQm1 * Hh * 2);
  int*   c_len  = (int*)take(Bb * 4);
  int*   qtok   = (int*)take(Bb * LQm1 * 4);
  int*   qmask  = (int*)take(Bb * LQm1 * 4);
  float* logZ   = (float*)take(Bb * LQm1 * 4);

  // prep
  k_prep_w<<<4 * Hh, 256, 0, stream>>>(eWhh, eWih, ebih, ebhh, Wcat_e, gb_e);
  k_prep_w<<<4 * Hh, 256, 0, stream>>>(dWhh, dWih, dbih, dbhh, Wcat_d, gb_d);
  k_prep_embw<<<Hh, 256, 0, stream>>>(embW, embWb);
  {
    long n8 = ((long)Vv * Hh) / 8;  // 3.2M
    k_cvt_bf16x8<<<(unsigned)((n8 + 255) / 256), 256, 0, stream>>>(pW, projWb, n8);
  }
  k_meta<<<1, 256, 0, stream>>>(cw, qw, c_len, qtok, qmask, h_a, h_b, c_st);

  // embeddings (context tokens are already flat [B*LC] in m = b*LC+t order)
  k_embed<<<(Bb * LCc) / 16, 256, 0, stream>>>(cw, wv, embWb, c_emb);
  k_embed<<<(Bb * LQm1) / 16, 256, 0, stream>>>(qtok, wv, embWb, q_emb);

  // encoder: 400 sequential steps, ping-pong h
  u16* hin = h_a; u16* hout = h_b;
  for (int t = 0; t < LCc; ++t) {
    k_lstm_step<<<Hh / 64, 256, 0, stream>>>(hin, hout, c_st, c_emb + (size_t)t * Hh,
                                             LCc * Hh, Wcat_e, gb_e, c_len, t, (u16*)nullptr);
    u16* tmp = hin; hin = hout; hout = tmp;
  }
  // decoder: 29 steps, record h into dec_o
  for (int t = 0; t < LQm1; ++t) {
    k_lstm_step<<<Hh / 64, 256, 0, stream>>>(hin, hout, c_st, q_emb + (size_t)t * Hh,
                                             LQm1 * Hh, Wcat_d, gb_d, (int*)nullptr, t, dec_o);
    u16* tmp = hin; hin = hout; hout = tmp;
  }

  // projection + masked log-softmax
  dim3 pg((Vv / 16 + 15) / 16, ((Bb * LQm1) / 16 + 3) / 4);  // (196, 15)
  k_proj<<<pg, 256, 0, stream>>>(dec_o, projWb, pb, qmask, out);
  k_lse<<<Bb * LQm1, 256, 0, stream>>>(out, logZ);
  dim3 ng((Vv / 4 + 255) / 256, Bb * LQm1);                  // (49, 928)
  k_norm<<<ng, 256, 0, stream>>>(out, logZ);
}
